// SequenceDecoder_429496730071
// MI455X (gfx1250) — matrix operand order repaired and checked by reading, not compiled, not run on hardware
//
#include <hip/hip_runtime.h>
#include <math.h>

// ---------------------------------------------------------------------------
// GRU sequence decoder, persistent-kernel formulation for gfx1250 (MI455X).
// B=4096, L=256, H=512, D=128, T=96.
//
// v2: M=32 batch tiles (128 WGs x 512 threads / 16 waves).  Each B-fragment
// (weights, streamed from L2) now feeds TWO WMMAs (two M-tiles), halving L2
// weight traffic vs the M=16 version.  fp32 hidden state lives in persistent
// registers (C-matrix layout); only the f16 copy (WMMA A operand shared by
// all waves) is kept in LDS.  All GEMMs: v_wmma_f32_16x16x32_f16, fp32 accum.
// ---------------------------------------------------------------------------

typedef _Float16 h16;
typedef __attribute__((ext_vector_type(16))) _Float16 v16h;
typedef __attribute__((ext_vector_type(8)))  _Float16 v8h;
typedef __attribute__((ext_vector_type(8)))  float    v8f;

union FragU { v16h v; struct { v8h lo, hi; } p; };

#define WMMA_F16(A, B, C) \
  __builtin_amdgcn_wmma_f32_16x16x32_f16(false, (A), false, (B), (short)0, (C), false, false)

// A-matrix fragment, 16x32 f16, row-major source with row stride `stride`.
// ISA 7.12.2: lane<16 -> M=lane, v0..3=K[k0..k0+7], v4..7=K[k0+16..k0+23];
// lane>=16 -> same M, K offset +8.
__device__ __forceinline__ v16h ldA(const h16* p0, int stride, int m, int k0, int half) {
  const h16* p = p0 + m * stride + k0 + half * 8;
  FragU f;
  f.p.lo = *(const v8h*)(p);
  f.p.hi = *(const v8h*)(p + 16);
  return f.v;
}

// B-matrix fragment, 32x16 f16 from W[n][k] row-major (B[k][n], contiguous k).
// B layout matches A: lane<16 -> N=lane, v0..3=K[k0..k0+7], v4..7=K[k0+16..k0+23]; lane>=16 -> K+8.
__device__ __forceinline__ v16h ldB(const h16* __restrict__ p0, int stride, int n, int k0, int half) {
  const h16* p = p0 + (size_t)n * stride + k0 + half * 8;
  FragU f;
  f.p.lo = *(const v8h*)(p);
  f.p.hi = *(const v8h*)(p + 16);
  return f.v;
}

__device__ __forceinline__ float fsig(float x)  { return 1.0f / (1.0f + __expf(-x)); }
__device__ __forceinline__ float ftanh(float x) { return 2.0f * fsig(2.0f * x) - 1.0f; }

#define ZV8 ((v8f){0.f,0.f,0.f,0.f,0.f,0.f,0.f,0.f})

// ---------------------------------------------------------------------------
// fp32 -> f16 weight conversion (prep pass; weights then live in 192MB L2)
// ---------------------------------------------------------------------------
__global__ void cvt_f32_to_f16(const float* __restrict__ src, h16* __restrict__ dst, int n) {
  int i = blockIdx.x * blockDim.x + threadIdx.x;
  if (i < n) dst[i] = (h16)src[i];
}

// ---------------------------------------------------------------------------
// Persistent GRU kernel: grid = 128 blocks (32 batch rows each), 512 threads.
// Wave w (0..15) owns h-columns [32w, 32w+32) of every gate slice, so r/z/n
// for a column land in the same wave's accumulators -> gate math is pure
// register work.
// ---------------------------------------------------------------------------
__global__ __launch_bounds__(512) void gru_decoder(
    const float* __restrict__ z0,     // (4096, 256)
    const float* __restrict__ x_init, // (4096, 1, 128)
    const h16*  __restrict__ Wlh,     // (512, 256)  f16
    const h16*  __restrict__ Wih,     // (1536, 128) f16
    const h16*  __restrict__ Whh,     // (1536, 512) f16
    const h16*  __restrict__ Wout,    // (128, 512)  f16
    const float* __restrict__ b_lh,   // (512)
    const float* __restrict__ b_ih,   // (1536)
    const float* __restrict__ b_hh,   // (1536)
    const float* __restrict__ b_out,  // (128)
    float* __restrict__ out)          // (4096, 96, 128)
{
  __shared__ __align__(16) h16 s_hb[32 * 512];  // h state, f16 (WMMA A, shared)
  __shared__ __align__(16) h16 s_x [32 * 128];  // x / y carry, f16 (WMMA A)
  __shared__ __align__(16) h16 s_z0[32 * 256];  // z0 tile, f16 (prologue A)
  __shared__ float s_bih[1536];
  __shared__ float s_bhh[1536];
  __shared__ float s_blh[512];
  __shared__ float s_bout[128];

  const int tid   = threadIdx.x;
  const int wave  = tid >> 5;       // 0..15
  const int lane  = tid & 31;
  const int half  = lane >> 4;      // 0/1 lane group
  const int idx   = lane & 15;      // row (A) / col (B,C) within 16-tile
  const int bbase = blockIdx.x * 32;
  const int colbase = wave * 32;    // this wave's h-columns [colbase, colbase+32)

  // ---- stage biases + convert z0 / x_init tiles to f16 in LDS -------------
  for (int i = tid; i < 1536; i += 512) { s_bih[i] = b_ih[i]; s_bhh[i] = b_hh[i]; }
  if (tid < 512) s_blh[tid] = b_lh[tid];
  if (tid < 128) s_bout[tid] = b_out[tid];
  for (int i = tid; i < 32 * 256; i += 512)
    s_z0[i] = (h16)z0[(size_t)(bbase + (i >> 8)) * 256 + (i & 255)];
  for (int i = tid; i < 32 * 128; i += 512)
    s_x[i] = (h16)x_init[(size_t)(bbase + (i >> 7)) * 128 + (i & 127)];
  __syncthreads();

  // Persistent fp32 hidden state, C-matrix layout: hreg[mt][nt][i] holds
  // h[bbase + mt*16 + i + 8*half][colbase + nt*16 + idx].
  v8f hreg[2][2];

  // ---- h0 = z0 @ W_lh^T + b_lh   (M=32, N=512, K=256) ---------------------
  {
    v8f acc[2][2];
#pragma unroll
    for (int mt = 0; mt < 2; ++mt)
#pragma unroll
      for (int nt = 0; nt < 2; ++nt) acc[mt][nt] = ZV8;

#pragma unroll 2
    for (int k = 0; k < 256; k += 32) {
      v16h a0 = ldA(s_z0, 256, idx,      k, half);
      v16h a1 = ldA(s_z0, 256, 16 + idx, k, half);
#pragma unroll
      for (int nt = 0; nt < 2; ++nt) {
        v16h b = ldB(Wlh, 256, colbase + nt * 16 + idx, k, half);
        acc[0][nt] = WMMA_F16(a0, b, acc[0][nt]);
        acc[1][nt] = WMMA_F16(a1, b, acc[1][nt]);
      }
    }
#pragma unroll
    for (int mt = 0; mt < 2; ++mt)
#pragma unroll
      for (int nt = 0; nt < 2; ++nt) {
        int j = colbase + nt * 16 + idx;
#pragma unroll
        for (int i = 0; i < 8; ++i) {
          int m = mt * 16 + i + 8 * half;
          float h = acc[mt][nt][i] + s_blh[j];
          hreg[mt][nt][i] = h;
          s_hb[m * 512 + j] = (h16)h;
        }
      }
  }
  __syncthreads();

  // ---- 96 recurrent steps --------------------------------------------------
  for (int t = 0; t < 96; ++t) {
    // acc tiles: [gate][mt][nt]; r,z get gi+gh summed; n kept split (ni, nh)
    v8f ar[2][2], az[2][2], ani[2][2], anh[2][2];
#pragma unroll
    for (int mt = 0; mt < 2; ++mt)
#pragma unroll
      for (int nt = 0; nt < 2; ++nt) {
        ar[mt][nt] = ZV8; az[mt][nt] = ZV8; ani[mt][nt] = ZV8; anh[mt][nt] = ZV8;
      }

    // gi = x @ W_ih^T : K = 128
#pragma unroll
    for (int k = 0; k < 128; k += 32) {
      v16h a0 = ldA(s_x, 128, idx,      k, half);
      v16h a1 = ldA(s_x, 128, 16 + idx, k, half);
#pragma unroll
      for (int nt = 0; nt < 2; ++nt) {
        int n = colbase + nt * 16 + idx;
        v16h br = ldB(Wih, 128, n,        k, half);
        ar[0][nt] = WMMA_F16(a0, br, ar[0][nt]);
        ar[1][nt] = WMMA_F16(a1, br, ar[1][nt]);
        v16h bz = ldB(Wih, 128, 512 + n,  k, half);
        az[0][nt] = WMMA_F16(a0, bz, az[0][nt]);
        az[1][nt] = WMMA_F16(a1, bz, az[1][nt]);
        v16h bn = ldB(Wih, 128, 1024 + n, k, half);
        ani[0][nt] = WMMA_F16(a0, bn, ani[0][nt]);
        ani[1][nt] = WMMA_F16(a1, bn, ani[1][nt]);
      }
    }

    // gh = h @ W_hh^T : K = 512 (r,z accumulate on top; n part -> anh)
#pragma unroll 4
    for (int k = 0; k < 512; k += 32) {
      v16h a0 = ldA(s_hb, 512, idx,      k, half);
      v16h a1 = ldA(s_hb, 512, 16 + idx, k, half);
#pragma unroll
      for (int nt = 0; nt < 2; ++nt) {
        int n = colbase + nt * 16 + idx;
        v16h br = ldB(Whh, 512, n,        k, half);
        ar[0][nt] = WMMA_F16(a0, br, ar[0][nt]);
        ar[1][nt] = WMMA_F16(a1, br, ar[1][nt]);
        v16h bz = ldB(Whh, 512, 512 + n,  k, half);
        az[0][nt] = WMMA_F16(a0, bz, az[0][nt]);
        az[1][nt] = WMMA_F16(a1, bz, az[1][nt]);
        v16h bn = ldB(Whh, 512, 1024 + n, k, half);
        anh[0][nt] = WMMA_F16(a0, bn, anh[0][nt]);
        anh[1][nt] = WMMA_F16(a1, bn, anh[1][nt]);
      }
    }
    __syncthreads();  // all waves done reading s_hb / s_x

    // Gate math: fully register-resident (h fp32 lives in hreg)
#pragma unroll
    for (int nt = 0; nt < 2; ++nt) {
      int j = colbase + nt * 16 + idx;
      float bir = s_bih[j],        bhr = s_bhh[j];
      float biz = s_bih[512 + j],  bhz = s_bhh[512 + j];
      float bin = s_bih[1024 + j], bhn = s_bhh[1024 + j];
#pragma unroll
      for (int mt = 0; mt < 2; ++mt) {
#pragma unroll
        for (int i = 0; i < 8; ++i) {
          int m = mt * 16 + i + 8 * half;
          float r  = fsig(ar[mt][nt][i] + bir + bhr);
          float zz = fsig(az[mt][nt][i] + biz + bhz);
          float nn = ftanh(ani[mt][nt][i] + bin + r * (anh[mt][nt][i] + bhn));
          float hnew = (1.0f - zz) * nn + zz * hreg[mt][nt][i];
          hreg[mt][nt][i] = hnew;
          s_hb[m * 512 + j] = (h16)hnew;
        }
      }
    }
    __syncthreads();  // h_new visible to all waves

    // y = h_new @ W_out^T + b_out : M=32, N=128 -> 16 C-tiles, one per wave.
    {
      const int ymt = wave >> 3;          // 0..1 : M-tile
      const int yn  = (wave & 7) * 16;    // N-tile base
      v8f ay = ZV8;
#pragma unroll 4
      for (int k = 0; k < 512; k += 32) {
        v16h a = ldA(s_hb, 512, ymt * 16 + idx, k, half);
        v16h b = ldB(Wout, 512, yn + idx,       k, half);
        ay = WMMA_F16(a, b, ay);
      }
      int d = yn + idx;
      float bo = s_bout[d];
#pragma unroll
      for (int i = 0; i < 8; ++i) {
        int m = ymt * 16 + i + 8 * half;
        float y = ay[i] + bo;
        out[((size_t)(bbase + m) * 96 + t) * 128 + d] = y;  // (B, T, D)
        s_x[m * 128 + d] = (h16)y;                          // carry for next gi
      }
    }
    __syncthreads();  // x ready for next step
  }
}

// ---------------------------------------------------------------------------
// Launch
// ---------------------------------------------------------------------------
extern "C" void kernel_launch(void* const* d_in, const int* in_sizes, int n_in,
                              void* d_out, int out_size, void* d_ws, size_t ws_size,
                              hipStream_t stream) {
  const float* z0     = (const float*)d_in[0];
  const float* x_init = (const float*)d_in[1];
  const float* W_lh   = (const float*)d_in[2];
  const float* b_lh   = (const float*)d_in[3];
  const float* W_ih   = (const float*)d_in[4];
  const float* W_hh   = (const float*)d_in[5];
  const float* b_ih   = (const float*)d_in[6];
  const float* b_hh   = (const float*)d_in[7];
  const float* W_out  = (const float*)d_in[8];
  const float* b_out  = (const float*)d_in[9];

  // f16 weight copies in workspace (2.36 MB total; resident in 192 MB L2)
  h16* Wlh_h  = (h16*)d_ws;
  h16* Wih_h  = Wlh_h + 512 * 256;
  h16* Whh_h  = Wih_h + 1536 * 128;
  h16* Wout_h = Whh_h + 1536 * 512;

  const int nlh  = 512 * 256;
  const int nih  = 1536 * 128;
  const int nhh  = 1536 * 512;
  const int nout = 128 * 512;
  cvt_f32_to_f16<<<(nlh  + 255) / 256, 256, 0, stream>>>(W_lh,  Wlh_h,  nlh);
  cvt_f32_to_f16<<<(nih  + 255) / 256, 256, 0, stream>>>(W_ih,  Wih_h,  nih);
  cvt_f32_to_f16<<<(nhh  + 255) / 256, 256, 0, stream>>>(W_hh,  Whh_h,  nhh);
  cvt_f32_to_f16<<<(nout + 255) / 256, 256, 0, stream>>>(W_out, Wout_h, nout);

  // 4096 batch rows / 32 per block = 128 persistent workgroups of 16 waves
  gru_decoder<<<128, 512, 0, stream>>>(z0, x_init, Wlh_h, Wih_h, Whh_h, Wout_h,
                                       b_lh, b_ih, b_hh, b_out, (float*)d_out);
}
